// CrissCrossAttention_87488483819612
// MI455X (gfx1250) — compile-verified, matrix operand order checked by reading, not run on hardware
//
#include <hip/hip_runtime.h>
#include <hip/hip_bf16.h>

typedef __attribute__((ext_vector_type(16))) __bf16 v16bf;
typedef __attribute__((ext_vector_type(8)))  float  v8f;
typedef __attribute__((ext_vector_type(4)))  unsigned v4u;
typedef __attribute__((ext_vector_type(8)))  int      v8i;
typedef __attribute__((ext_vector_type(4)))  int      v4i;

constexpr int NB  = 4;
constexpr int CIN = 512;
constexpr int CQ  = 64;
constexpr int H   = 97;
constexpr int W   = 97;
constexpr int HW  = H * W;          // 9409
constexpr int KP  = 112;            // 97 padded to 7 tiles of 16
constexpr int NT  = 7;              // tiles along 97

__device__ __forceinline__ int kmap(int e, int hi) {
  // V_WMMA_F32_16X16X32_BF16 A/B fragment K index for element e, lane-half hi
  return ((e & 8) << 1) + (hi << 3) + (e & 7);
}

__device__ __forceinline__ v8f wmma_bf16(v16bf a, v16bf b, v8f c) {
  return __builtin_amdgcn_wmma_f32_16x16x32_bf16(false, a, false, b, (short)0, c,
                                                 false, false);
}

// Low 32 bits of a flat pointer into __shared__ == LDS byte offset (addrspace(3)).
__device__ __forceinline__ unsigned lds_off(const void* p) {
  return (unsigned)(size_t)p;
}

// Async global->LDS copy, 4 bytes per active lane; per-lane LDS dest + global offset.
__device__ __forceinline__ void async_copy_b32(unsigned lds, const float* base,
                                               unsigned byte_off) {
  asm volatile("global_load_async_to_lds_b32 %0, %1, %2"
               :: "v"(lds), "v"(byte_off), "s"(base)
               : "memory");
}

__device__ __forceinline__ void wait_async() {
  asm volatile("s_wait_asynccnt 0" ::: "memory");
}

// TDM: DMA a 2D tile (rows x cols f32, row stride in elements) global -> LDS.
// D# built per CDNA5 ISA ch.8: group0 {count=1, lds_addr, global_addr, type=2},
// group1 {data_size=4B, tensor_dim0=cols, tensor_dim1=rows, tile=cols x rows,
// tensor_dim0_stride=row_stride}; groups 2/3 zero (2D tensor).
__device__ __forceinline__ void tdm_load_2d(unsigned lds, const float* gptr,
                                            int cols, int rows, int row_stride) {
  const unsigned long long ga = (unsigned long long)gptr;
  v4u g0 = { 1u,                                   // count=1, user desc
             lds,                                  // lds_addr (bytes)
             (unsigned)(ga & 0xffffffffu),
             (unsigned)((ga >> 32) & 0x01ffffffu) | 0x80000000u }; // type=2
  v8i g1;
  g1[0] = (2 << 16);                               // data_size: 4 bytes
  g1[1] = (cols & 0xffff) << 16;                   // tensor_dim0[15:0]
  g1[2] = ((cols >> 16) & 0xffff) | ((rows & 0xffff) << 16); // tdim0 hi | tdim1 lo
  g1[3] = ((rows >> 16) & 0xffff) | ((cols & 0xffff) << 16); // tdim1 hi | tile_dim0
  g1[4] = (rows & 0xffff);                         // tile_dim1 | tile_dim2=0
  g1[5] = row_stride;                              // tensor_dim0_stride[31:0]
  g1[6] = 0;                                       // stride hi | tensor_dim1_stride lo
  g1[7] = 0;
  v4i gz = { 0, 0, 0, 0 };
  asm volatile("tensor_load_to_lds %0, %1, %2, %3"
               :: "s"(g0), "s"(g1), "s"(gz), "s"(gz)
               : "memory");
}

// ---------------------------------------------------------------------------
// 1) Projections: out[n,m,p] = sum_c w[m,c] * x[n,c,p] + bias[m]
//    One wave per 16(out-ch) x 64(pixel) tile: A fragment (weights) reused
//    across 4 pixel sub-tiles; K = 512 in steps of 32 (bf16).
// ---------------------------------------------------------------------------
__global__ void proj_kernel(const float* __restrict__ x, const float* __restrict__ w,
                            const float* __restrict__ bias, float* __restrict__ out,
                            int Cout) {
  const int lane = threadIdx.x & 31, l = lane & 15, hi = lane >> 4;
  const int pt = blockIdx.x << 6;         // 64-pixel tile base
  const int m0 = blockIdx.y << 4;         // out-channel tile base
  const int n  = blockIdx.z;
  const float* xb = x + (size_t)n * CIN * HW;
  int p[4], pc[4];
#pragma unroll
  for (int j = 0; j < 4; ++j) {
    p[j]  = pt + (j << 4) + l;
    pc[j] = (p[j] < HW) ? p[j] : (HW - 1);   // clamp loads; stores guarded
  }
  v8f acc[4] = {{}, {}, {}, {}};
  for (int kk = 0; kk < CIN; kk += 32) {
    if (kk + 32 < CIN)
      __builtin_prefetch(&xb[(size_t)(kk + 32 + (hi << 3)) * HW + pc[0]], 0, 1);
    v16bf a;
#pragma unroll
    for (int e = 0; e < 16; ++e)
      a[e] = (__bf16)w[(size_t)(m0 + l) * CIN + kk + kmap(e, hi)];
#pragma unroll
    for (int j = 0; j < 4; ++j) {
      v16bf b;
#pragma unroll
      for (int e = 0; e < 16; ++e)
        b[e] = (__bf16)xb[(size_t)(kk + kmap(e, hi)) * HW + pc[j]];
      acc[j] = wmma_bf16(a, b, acc[j]);
    }
  }
#pragma unroll
  for (int j = 0; j < 4; ++j) {
#pragma unroll
    for (int d = 0; d < 8; ++d) {
      const int m = m0 + d + (hi << 3);
      if (p[j] < HW)
        out[(size_t)n * Cout * HW + (size_t)m * HW + p[j]] = acc[j][d] + bias[m];
    }
  }
}

// ---------------------------------------------------------------------------
// 2) Row energies per (n,y):  E[x,i] = sum_c q[c,y,x] * k[c,y,i]
//    q/k slabs (64 rows x 97 f32, stride HW) DMA'd into LDS by the TDM.
//    No pad fill: garbage in columns >=97 only reaches guarded-out stores.
// ---------------------------------------------------------------------------
__global__ void row_energy_kernel(const float* __restrict__ q,
                                  const float* __restrict__ k,
                                  float* __restrict__ erow) {
  __shared__ float qs[CQ * W + 32];
  __shared__ float ks[CQ * W + 32];
  const int n = blockIdx.x / H, y = blockIdx.x % H;
  const int tid = threadIdx.x;
  if ((tid >> 5) == 0) {
    const float* qbase = q + (size_t)n * CQ * HW + (size_t)y * W;
    const float* kbase = k + (size_t)n * CQ * HW + (size_t)y * W;
    tdm_load_2d(lds_off(qs), qbase, W, CQ, HW);
    tdm_load_2d(lds_off(ks), kbase, W, CQ, HW);
    __builtin_amdgcn_s_wait_tensorcnt(0);
  }
  __syncthreads();
  const int wave = tid >> 5, lane = tid & 31, l = lane & 15, hi = lane >> 4;
  float* er = erow + (size_t)(n * H + y) * W * W;
  for (int t = wave; t < NT * NT; t += (int)(blockDim.x >> 5)) {
    const int m0 = (t / NT) << 4, n0 = (t % NT) << 4;
    v8f acc = {};
#pragma unroll
    for (int kk = 0; kk < CQ; kk += 32) {
      v16bf a, b;
#pragma unroll
      for (int e = 0; e < 16; ++e) {
        const int kc = kk + kmap(e, hi);
        a[e] = (__bf16)qs[kc * W + m0 + l];
        b[e] = (__bf16)ks[kc * W + n0 + l];
      }
      acc = wmma_bf16(a, b, acc);
    }
#pragma unroll
    for (int d = 0; d < 8; ++d) {
      const int xq = m0 + d + (hi << 3), i = n0 + l;
      if (xq < W && i < W) er[(size_t)xq * W + i] = acc[d];
    }
  }
}

// ---------------------------------------------------------------------------
// 3) Column energies per (n,x):  E[y,i] = sum_c q[c,y,x] * k[c,i,x]
//    Strided gather staged via GLOBAL_LOAD_ASYNC_TO_LDS_B32 (valid region only).
// ---------------------------------------------------------------------------
__global__ void col_energy_kernel(const float* __restrict__ q,
                                  const float* __restrict__ k,
                                  float* __restrict__ ecol) {
  __shared__ float qs[CQ * W + 32];
  __shared__ float ks[CQ * W + 32];
  const int n = blockIdx.x / W, x = blockIdx.x % W;
  const int tid = threadIdx.x;
  const float* qbase = q + (size_t)n * CQ * HW + (size_t)x;
  const float* kbase = k + (size_t)n * CQ * HW + (size_t)x;
  for (int idx = tid; idx < CQ * W; idx += blockDim.x) {
    const int c = idx / W, pp = idx % W;
    const unsigned off = (unsigned)((c * HW + pp * W) * 4);
    async_copy_b32(lds_off(&qs[idx]), qbase, off);
    async_copy_b32(lds_off(&ks[idx]), kbase, off);
  }
  wait_async();
  __syncthreads();
  const int wave = tid >> 5, lane = tid & 31, l = lane & 15, hi = lane >> 4;
  float* ec = ecol + (size_t)(n * W + x) * H * H;
  for (int t = wave; t < NT * NT; t += (int)(blockDim.x >> 5)) {
    const int m0 = (t / NT) << 4, n0 = (t % NT) << 4;
    v8f acc = {};
#pragma unroll
    for (int kk = 0; kk < CQ; kk += 32) {
      v16bf a, b;
#pragma unroll
      for (int e = 0; e < 16; ++e) {
        const int kc = kk + kmap(e, hi);
        a[e] = (__bf16)qs[kc * W + m0 + l];
        b[e] = (__bf16)ks[kc * W + n0 + l];
      }
      acc = wmma_bf16(a, b, acc);
    }
#pragma unroll
    for (int d = 0; d < 8; ++d) {
      const int yq = m0 + d + (hi << 3), i = n0 + l;
      if (yq < H && i < H) ec[(size_t)yq * H + i] = acc[d];
    }
  }
}

// ---------------------------------------------------------------------------
// 4) Softmax over concat(row[97], col[i != y]) per pixel; one wave per pixel.
// ---------------------------------------------------------------------------
__global__ void softmax_kernel(float* __restrict__ erow, float* __restrict__ ecol) {
  const int wid  = (int)((blockIdx.x * blockDim.x + threadIdx.x) >> 5);
  const int lane = threadIdx.x & 31;
  if (wid >= NB * H * W) return;
  const int n = wid / HW, rem = wid % HW, y = rem / W, x = rem % W;
  float* rb = erow + ((size_t)(n * H + y) * W + x) * (size_t)W;
  float* cb = ecol + ((size_t)(n * W + x) * H + y) * (size_t)H;

  float rv[4], cv[4];
  float mx = -1e30f;
#pragma unroll
  for (int t = 0; t < 4; ++t) {
    const int i = lane + t * 32;
    float r = (i < W) ? rb[i] : -1e30f;
    float c = (i < H && i != y) ? cb[i] : -1e30f;
    rv[t] = r; cv[t] = c;
    mx = fmaxf(mx, fmaxf(r, c));
  }
#pragma unroll
  for (int off = 16; off > 0; off >>= 1) mx = fmaxf(mx, __shfl_xor(mx, off, 32));
  float s = 0.f;
#pragma unroll
  for (int t = 0; t < 4; ++t) {
    rv[t] = (rv[t] > -1e29f) ? __expf(rv[t] - mx) : 0.f;
    cv[t] = (cv[t] > -1e29f) ? __expf(cv[t] - mx) : 0.f;
    s += rv[t] + cv[t];
  }
#pragma unroll
  for (int off = 16; off > 0; off >>= 1) s += __shfl_xor(s, off, 32);
  const float inv = 1.0f / s;
#pragma unroll
  for (int t = 0; t < 4; ++t) {
    const int i = lane + t * 32;
    if (i < W) rb[i] = rv[t] * inv;
    if (i < H) cb[i] = (i == y) ? 0.f : cv[t] * inv;
  }
}

// ---------------------------------------------------------------------------
// 5) Column aggregation per (n,x): colpart[c,y] = sum_i a_col[y,i] * v[c,i,x]
//    Written to d_out (as a partial). K pad (i>=97) must be zero-filled here.
// ---------------------------------------------------------------------------
__global__ void col_agg_kernel(const float* __restrict__ v,
                               const float* __restrict__ acol,
                               float* __restrict__ outp) {
  __shared__ __bf16 at[128 * KP];   // [K=i up to 128][y up to 112]
  __shared__ __bf16 vs[128 * 128];  // [c-chunk 128][K=i up to 128]
  const int n = blockIdx.x / W, x = blockIdx.x % W;
  const int tid = threadIdx.x;
  const float* ac = acol + (size_t)(n * W + x) * H * H;
  for (int idx = tid; idx < 128 * KP; idx += blockDim.x) {
    const int i = idx & 127, yq = idx >> 7;
    float a = 0.f;
    if (yq < H && i < H) a = ac[(size_t)yq * H + i];
    at[i * KP + yq] = (__bf16)a;
  }
  const size_t vb = (size_t)n * CIN * HW + (size_t)x;
  const int wave = tid >> 5, lane = tid & 31, l = lane & 15, hi = lane >> 4;
  for (int c0 = 0; c0 < CIN; c0 += 128) {
    __syncthreads();
    for (int idx = tid; idx < 128 * 128; idx += blockDim.x) {
      const int i = idx & 127, c = idx >> 7;
      float vv = 0.f;
      if (i < H) vv = v[vb + (size_t)(c0 + c) * HW + (size_t)i * W];
      vs[idx] = (__bf16)vv;
    }
    __syncthreads();
    for (int t = wave; t < 8 * NT; t += (int)(blockDim.x >> 5)) {
      const int m0 = (t / NT) << 4, n0 = (t % NT) << 4;
      v8f acc = {};
#pragma unroll
      for (int kk = 0; kk < 128; kk += 32) {
        v16bf a, b;
#pragma unroll
        for (int e = 0; e < 16; ++e) {
          const int kc = kk + kmap(e, hi);
          a[e] = vs[((m0 + l) << 7) + kc];
          b[e] = at[kc * KP + n0 + l];
        }
        acc = wmma_bf16(a, b, acc);
      }
#pragma unroll
      for (int d = 0; d < 8; ++d) {
        const int c = c0 + m0 + d + (hi << 3), yq = n0 + l;
        if (yq < H)
          outp[(size_t)n * CIN * HW + (size_t)c * HW + (size_t)yq * W + x] = acc[d];
      }
    }
  }
}

// ---------------------------------------------------------------------------
// 6) Row aggregation + combine per (n,y):
//    out[c,x] = gamma * ( sum_i a_row[x,i]*v[c,y,i] + colpart[c,x] ) + xin[c,x]
// ---------------------------------------------------------------------------
__global__ void row_agg_kernel(const float* __restrict__ v,
                               const float* __restrict__ arow,
                               const float* __restrict__ xin,
                               const float* __restrict__ gamma,
                               float* __restrict__ outp) {
  __shared__ __bf16 at[128 * KP];   // [K=i up to 128][x up to 112]
  __shared__ __bf16 vs[128 * 128];  // [c-chunk 128][K=i up to 128]
  const int n = blockIdx.x / H, y = blockIdx.x % H;
  const int tid = threadIdx.x;
  const float g = gamma[0];
  const float* ar = arow + (size_t)(n * H + y) * W * W;
  for (int idx = tid; idx < 128 * KP; idx += blockDim.x) {
    const int i = idx & 127, xq = idx >> 7;
    float a = 0.f;
    if (xq < W && i < W) a = ar[(size_t)xq * W + i];
    at[i * KP + xq] = (__bf16)a;
  }
  const size_t vb = (size_t)n * CIN * HW + (size_t)y * W;
  const int wave = tid >> 5, lane = tid & 31, l = lane & 15, hi = lane >> 4;
  for (int c0 = 0; c0 < CIN; c0 += 128) {
    __syncthreads();
    for (int idx = tid; idx < 128 * 128; idx += blockDim.x) {
      const int i = idx & 127, c = idx >> 7;
      float vv = 0.f;
      if (i < W) vv = v[vb + (size_t)(c0 + c) * HW + i];
      vs[idx] = (__bf16)vv;
    }
    __syncthreads();
    for (int t = wave; t < 8 * NT; t += (int)(blockDim.x >> 5)) {
      const int m0 = (t / NT) << 4, n0 = (t % NT) << 4;
      v8f acc = {};
#pragma unroll
      for (int kk = 0; kk < 128; kk += 32) {
        v16bf a, b;
#pragma unroll
        for (int e = 0; e < 16; ++e) {
          const int kc = kk + kmap(e, hi);
          a[e] = vs[((m0 + l) << 7) + kc];
          b[e] = at[kc * KP + n0 + l];
        }
        acc = wmma_bf16(a, b, acc);
      }
#pragma unroll
      for (int d = 0; d < 8; ++d) {
        const int c = c0 + m0 + d + (hi << 3), xq = n0 + l;
        if (xq < W) {
          const size_t o = (size_t)n * CIN * HW + (size_t)c * HW + (size_t)y * W + xq;
          outp[o] = g * (acc[d] + outp[o]) + xin[o];
        }
      }
    }
  }
}

// ---------------------------------------------------------------------------
extern "C" void kernel_launch(void* const* d_in, const int* in_sizes, int n_in,
                              void* d_out, int out_size, void* d_ws, size_t ws_size,
                              hipStream_t stream) {
  (void)in_sizes; (void)n_in; (void)out_size; (void)ws_size;
  const float* x     = (const float*)d_in[0];
  const float* wq    = (const float*)d_in[1];
  const float* bq    = (const float*)d_in[2];
  const float* wk    = (const float*)d_in[3];
  const float* bk    = (const float*)d_in[4];
  const float* wv    = (const float*)d_in[5];
  const float* bv    = (const float*)d_in[6];
  const float* gamma = (const float*)d_in[7];
  float* out = (float*)d_out;

  float* wsf = (float*)d_ws;
  const size_t QSZ = (size_t)NB * CQ * HW;    // 2,408,704 f
  const size_t VSZ = (size_t)NB * CIN * HW;   // 19,269,632 f
  const size_t ESZ = (size_t)NB * H * W * W;  // 3,650,692 f
  float* qb   = wsf;
  float* kb   = qb + QSZ;
  float* vb   = kb + QSZ;
  float* erow = vb + VSZ;
  float* ecol = erow + ESZ;

  const dim3 bProj(32, 1, 1);
  const int ptiles = (HW + 63) / 64;          // 148 (64 pixels per wave)
  proj_kernel<<<dim3(ptiles, CQ / 16, NB),  bProj, 0, stream>>>(x, wq, bq, qb, CQ);
  proj_kernel<<<dim3(ptiles, CQ / 16, NB),  bProj, 0, stream>>>(x, wk, bk, kb, CQ);
  proj_kernel<<<dim3(ptiles, CIN / 16, NB), bProj, 0, stream>>>(x, wv, bv, vb, CIN);

  row_energy_kernel<<<dim3(NB * H), dim3(256), 0, stream>>>(qb, kb, erow);
  col_energy_kernel<<<dim3(NB * W), dim3(256), 0, stream>>>(qb, kb, ecol);

  const int npix = NB * H * W;                       // 37,636 pixels (1 wave each)
  softmax_kernel<<<dim3((npix + 7) / 8), dim3(256), 0, stream>>>(erow, ecol);

  col_agg_kernel<<<dim3(NB * W), dim3(256), 0, stream>>>(vb, ecol, out);
  row_agg_kernel<<<dim3(NB * H), dim3(256), 0, stream>>>(vb, erow, x, gamma, out);
}
